// MultiHeadAttention_78194174591517
// MI455X (gfx1250) — compile-verified, matrix operand order checked by reading, not run on hardware
//
#include <hip/hip_runtime.h>
#include <hip/hip_bf16.h>

typedef __attribute__((ext_vector_type(16))) __bf16 bf16x16;
typedef __attribute__((ext_vector_type(8)))  float  f32x8;
typedef unsigned int v4u __attribute__((ext_vector_type(4)));
typedef int          v8i __attribute__((ext_vector_type(8)));
typedef int          v4i __attribute__((ext_vector_type(4)));

#define S_LEN   4096
#define D_MODEL 768
#define N_HEADS 12
#define DEPTH   64
#define KVPITCH 72          // 64 elems + 4-DWORD TDM pad -> 144B rows, bank-friendly

// ---------------------------------------------------------------------------
// Fragment load: 16 bf16 per lane as two 16-byte chunks (K and K+16), matching
// the CDNA5 16-bit A/B fragment layout (lane half selects K+0 / K+8 subrows).
// ---------------------------------------------------------------------------
__device__ inline bf16x16 frag_ld(const __bf16* p) {
    union { uint4 u[2]; bf16x16 v; } f;
    f.u[0] = *reinterpret_cast<const uint4*>(p);
    f.u[1] = *reinterpret_cast<const uint4*>(p + 16);
    return f.v;
}

__device__ inline f32x8 wmma_bf16(bf16x16 a, bf16x16 b, f32x8 c) {
    return __builtin_amdgcn_wmma_f32_16x16x32_bf16(
        false, a, false, b, (short)0, c, false, false);
}

// ---------------------------------------------------------------------------
// Tensor Data Mover: async-load one 64x64 bf16 tile (row stride = rowStride
// elements) into LDS at lds_off, padding +4 DWORDs every 32 DWORDs so the LDS
// row pitch becomes 72 elements (144 B) for conflict-free ds_load_b128.
// D# packing per CDNA5 ISA (groups 2/3 zero => 2D tensor).
// 6-arg builtin form (clang-23 / therock-10.0 lane).
// ---------------------------------------------------------------------------
__device__ inline void tdm_load_tile64(const __bf16* gsrc, unsigned lds_off,
                                       unsigned long long rowStride) {
    const unsigned long long ga = (unsigned long long)(size_t)gsrc;
    v4u g0;
    g0[0] = 1u;                                   // count=1 (valid user D#)
    g0[1] = lds_off;                              // lds_addr  [63:32]
    g0[2] = (unsigned)ga;                         // global_addr[31:0]
    g0[3] = (unsigned)((ga >> 32) & 0x01FFFFFFu)  // global_addr[56:32]
          | (2u << 30);                           // type=2 ("image")
    v8i g1;
    g1[0] = (int)((1u << 16)                      // data_size = 2 bytes
                | (1u << 20)                      // pad_enable
                | (4u << 22)                      // pad_interval: 32 DWORDs
                | (3u << 25));                    // pad_amount : 4 DWORDs
    g1[1] = (int)(64u << 16);                     // tensor_dim0 = 64
    g1[2] = (int)(64u << 16);                     // tensor_dim1 = 64 (lo)
    g1[3] = (int)(64u << 16);                     // tile_dim0   = 64
    g1[4] = (int)64;                              // tile_dim1   = 64
    g1[5] = (int)(unsigned)rowStride;             // tensor_dim0_stride lo
    g1[6] = (int)(unsigned)(rowStride >> 32);     // tensor_dim0_stride hi
    g1[7] = 0;
    v4i z4 = {};
    v8i z8 = {};
    __builtin_amdgcn_tensor_load_to_lds(g0, g1, z4, z4, z8, 0);
}

// ---------------------------------------------------------------------------
// f32 -> bf16 conversion
// ---------------------------------------------------------------------------
__global__ void cvt_f32_bf16(const float* __restrict__ in,
                             __bf16* __restrict__ out, int n) {
    int i = blockIdx.x * blockDim.x + threadIdx.x;
    if (i < n) out[i] = (__bf16)in[i];
}

// ---------------------------------------------------------------------------
// C[M,N] = A[M,K] * Bsrc[N,K]^T   (both row-major, K contiguous)
// Each wave computes a 64x64 tile: 4x4 grid of 16x16 accumulators.
// flags bit0: f32 output, bit1: bias indexed by row (for transposed outputs)
// ---------------------------------------------------------------------------
__global__ void __launch_bounds__(256)
gemm_bf16_nt(const __bf16* __restrict__ A, const __bf16* __restrict__ Bsrc,
             void* __restrict__ Cout, const float* __restrict__ bias,
             int M, int N, int K, int flags) {
    const int lane = threadIdx.x & 31;
    const int wid  = threadIdx.x >> 5;
    const int gw   = blockIdx.x * 8 + wid;
    const int ntiles = N >> 6;
    const int total  = (M >> 6) * ntiles;
    if (gw >= total) return;
    const int tm = gw / ntiles, tn = gw % ntiles;
    const int m0 = tm << 6, n0 = tn << 6;
    const int l15 = lane & 15;
    const int h8  = (lane >> 4) << 3;

    f32x8 acc[4][4] = {};

    for (int k0 = 0; k0 < K; k0 += 32) {
        bf16x16 af[4], bfm[4];
#pragma unroll
        for (int i = 0; i < 4; ++i)
            af[i] = frag_ld(A + (size_t)(m0 + i * 16 + l15) * K + k0 + h8);
#pragma unroll
        for (int j = 0; j < 4; ++j)
            bfm[j] = frag_ld(Bsrc + (size_t)(n0 + j * 16 + l15) * K + k0 + h8);
        if (k0 + 32 < K) {
            __builtin_prefetch(A + (size_t)(m0 + l15) * K + k0 + 32, 0, 3);
            __builtin_prefetch(Bsrc + (size_t)(n0 + l15) * K + k0 + 32, 0, 3);
        }
#pragma unroll
        for (int i = 0; i < 4; ++i)
#pragma unroll
            for (int j = 0; j < 4; ++j)
                acc[i][j] = wmma_bf16(af[i], bfm[j], acc[i][j]);
    }

    const bool f32out  = (flags & 1) != 0;
    const bool biasRow = (flags & 2) != 0;
#pragma unroll
    for (int i = 0; i < 4; ++i)
#pragma unroll
        for (int j = 0; j < 4; ++j) {
            const int col = n0 + j * 16 + l15;
#pragma unroll
            for (int r = 0; r < 8; ++r) {
                const int row = m0 + i * 16 + r + h8;
                float v = acc[i][j][r];
                if (bias) v += biasRow ? bias[row] : bias[col];
                if (f32out)
                    ((float*)Cout)[(size_t)row * N + col] = v;
                else
                    ((__bf16*)Cout)[(size_t)row * N + col] = (__bf16)v;
            }
        }
}

// ---------------------------------------------------------------------------
// Flash attention. Block = 8 waves = 256 queries of ONE head; each wave owns
// 32 queries. K/V 64-key tiles are shared by all 8 waves through LDS, loaded
// by the Tensor Data Mover (double-buffered, TENSORcnt-tracked) while WMMAs
// run on the current buffer. Q,K row-major [S,768]; V transposed [768,S].
// ---------------------------------------------------------------------------
__global__ void __launch_bounds__(256)
attn_flash(const __bf16* __restrict__ Q, const __bf16* __restrict__ Kmat,
           const __bf16* __restrict__ Vt, __bf16* __restrict__ ctx) {
    __shared__ __bf16 kvt[2][2][64 * KVPITCH];   // [buf][K/V][tile] via TDM
    __shared__ __bf16 pstage[8][32 * 72];        // per-wave P re-layout

    const int lane = threadIdx.x & 31;
    const int wid  = threadIdx.x >> 5;
    const int h    = blockIdx.x >> 4;            // 16 blocks per head
    const int q0   = ((blockIdx.x & 15) << 8) + (wid << 5);
    const int e0   = h * DEPTH;
    const int l15  = lane & 15;
    const int h8   = (lane >> 4) << 3;
    const float scale = 0.125f;                  // 1/sqrt(64)

    // Q fragments: fixed for the whole key loop (2 m-tiles x 2 k-steps)
    bf16x16 qf[2][2];
#pragma unroll
    for (int mt = 0; mt < 2; ++mt)
#pragma unroll
        for (int ks = 0; ks < 2; ++ks)
            qf[mt][ks] = frag_ld(Q + (size_t)(q0 + mt * 16 + l15) * D_MODEL +
                                 e0 + ks * 32 + h8);

    // Prologue: TDM-load first K/V tiles into buffer 0
    if (wid == 0) {
        tdm_load_tile64(Kmat + (size_t)0 * D_MODEL + e0,
                        (unsigned)(size_t)&kvt[0][0][0], D_MODEL);
        tdm_load_tile64(Vt + (size_t)e0 * S_LEN + 0,
                        (unsigned)(size_t)&kvt[0][1][0], S_LEN);
        __builtin_amdgcn_s_wait_tensorcnt(0);
    }
    __syncthreads();

    f32x8 cacc[2][4] = {};
    float mrun[2][8], lrun[2][8];
#pragma unroll
    for (int mt = 0; mt < 2; ++mt)
#pragma unroll
        for (int r = 0; r < 8; ++r) { mrun[mt][r] = -1e30f; lrun[mt][r] = 0.f; }

    int buf = 0;
    for (int kb = 0; kb < S_LEN; kb += 64) {
        const bool more = (kb + 64) < S_LEN;
        // Issue next tiles into the other buffer; overlap with compute below.
        if (wid == 0 && more) {
            tdm_load_tile64(Kmat + (size_t)(kb + 64) * D_MODEL + e0,
                            (unsigned)(size_t)&kvt[buf ^ 1][0][0], D_MODEL);
            tdm_load_tile64(Vt + (size_t)e0 * S_LEN + (kb + 64),
                            (unsigned)(size_t)&kvt[buf ^ 1][1][0], S_LEN);
        }
        const __bf16* kt = &kvt[buf][0][0];
        const __bf16* vt = &kvt[buf][1][0];

        // S_tile[32x64] = Q_blk * K_blk^T  (K-dim = depth 64 -> 2 wmma steps)
        f32x8 sc[2][4];
#pragma unroll
        for (int nt = 0; nt < 4; ++nt) {
            bf16x16 kf0 = frag_ld(kt + (nt * 16 + l15) * KVPITCH + h8);
            bf16x16 kf1 = frag_ld(kt + (nt * 16 + l15) * KVPITCH + 32 + h8);
#pragma unroll
            for (int mt = 0; mt < 2; ++mt) {
                f32x8 c = {};
                c = wmma_bf16(qf[mt][0], kf0, c);
                c = wmma_bf16(qf[mt][1], kf1, c);
                sc[mt][nt] = c;
            }
        }

        // Online softmax. C layout: lane half holds rows r / r+8, cols in lanes.
#pragma unroll
        for (int mt = 0; mt < 2; ++mt) {
#pragma unroll
            for (int r = 0; r < 8; ++r) {
                float mx = -1e30f;
#pragma unroll
                for (int nt = 0; nt < 4; ++nt) mx = fmaxf(mx, sc[mt][nt][r]);
                mx *= scale;
#pragma unroll
                for (int off = 1; off < 16; off <<= 1)
                    mx = fmaxf(mx, __shfl_xor(mx, off, 32));
                const float mnew  = fmaxf(mrun[mt][r], mx);
                const float alpha = __expf(mrun[mt][r] - mnew);
                float rs = 0.f;
#pragma unroll
                for (int nt = 0; nt < 4; ++nt) {
                    const float p = __expf(sc[mt][nt][r] * scale - mnew);
                    rs += p;
                    pstage[wid][(mt * 16 + r + h8) * 72 + nt * 16 + l15] = (__bf16)p;
                }
#pragma unroll
                for (int off = 1; off < 16; off <<= 1)
                    rs += __shfl_xor(rs, off, 32);
                lrun[mt][r] = lrun[mt][r] * alpha + rs;
                mrun[mt][r] = mnew;
#pragma unroll
                for (int dt = 0; dt < 4; ++dt) cacc[mt][dt][r] *= alpha;
            }
        }

        __syncthreads();
        // Re-shape P (C-layout in LDS) into A-layout fragments
        bf16x16 pf[2][2];
#pragma unroll
        for (int mt = 0; mt < 2; ++mt)
#pragma unroll
            for (int ks = 0; ks < 2; ++ks)
                pf[mt][ks] = frag_ld(&pstage[wid][(mt * 16 + l15) * 72 + ks * 32 + h8]);
        __syncthreads();

        // ctx += P * V_blk  (B columns = rows of Vt tile, contiguous keys)
#pragma unroll
        for (int dt = 0; dt < 4; ++dt) {
            bf16x16 vf0 = frag_ld(vt + (dt * 16 + l15) * KVPITCH + h8);
            bf16x16 vf1 = frag_ld(vt + (dt * 16 + l15) * KVPITCH + 32 + h8);
#pragma unroll
            for (int mt = 0; mt < 2; ++mt) {
                cacc[mt][dt] = wmma_bf16(pf[mt][0], vf0, cacc[mt][dt]);
                cacc[mt][dt] = wmma_bf16(pf[mt][1], vf1, cacc[mt][dt]);
            }
        }

        // Drain TDM for the next buffer, publish to all waves, swap.
        if (wid == 0 && more) __builtin_amdgcn_s_wait_tensorcnt(0);
        __syncthreads();
        buf ^= 1;
    }

    // Normalize and store ctx (bf16, [S, 768])
#pragma unroll
    for (int mt = 0; mt < 2; ++mt)
#pragma unroll
        for (int dt = 0; dt < 4; ++dt)
#pragma unroll
            for (int r = 0; r < 8; ++r) {
                const float v = cacc[mt][dt][r] / lrun[mt][r];
                ctx[(size_t)(q0 + mt * 16 + r + h8) * D_MODEL +
                    e0 + dt * 16 + l15] = (__bf16)v;
            }
}

// ---------------------------------------------------------------------------
extern "C" void kernel_launch(void* const* d_in, const int* in_sizes, int n_in,
                              void* d_out, int out_size, void* d_ws, size_t ws_size,
                              hipStream_t stream) {
    (void)in_sizes; (void)n_in; (void)out_size; (void)ws_size;

    const float* x    = (const float*)d_in[0];
    const float* wq_w = (const float*)d_in[1];
    const float* wq_b = (const float*)d_in[2];
    const float* wk_w = (const float*)d_in[3];
    const float* wk_b = (const float*)d_in[4];
    const float* wv_w = (const float*)d_in[5];
    const float* wv_b = (const float*)d_in[6];
    const float* wo_w = (const float*)d_in[7];
    const float* wo_b = (const float*)d_in[8];

    const size_t nx = (size_t)S_LEN * D_MODEL;       // 3,145,728
    const size_t nw = (size_t)D_MODEL * D_MODEL;     //   589,824

    __bf16* xb   = (__bf16*)d_ws;
    __bf16* wqb  = xb   + nx;
    __bf16* wkb  = wqb  + nw;
    __bf16* wvb  = wkb  + nw;
    __bf16* wob  = wvb  + nw;
    __bf16* Qb   = wob  + nw;
    __bf16* Kb   = Qb   + nx;
    __bf16* Vtb  = Kb   + nx;
    __bf16* ctxb = Vtb  + nx;

    // 1) convert everything to bf16
    cvt_f32_bf16<<<(int)((nx + 255) / 256), 256, 0, stream>>>(x, xb, (int)nx);
    cvt_f32_bf16<<<(int)((nw + 255) / 256), 256, 0, stream>>>(wq_w, wqb, (int)nw);
    cvt_f32_bf16<<<(int)((nw + 255) / 256), 256, 0, stream>>>(wk_w, wkb, (int)nw);
    cvt_f32_bf16<<<(int)((nw + 255) / 256), 256, 0, stream>>>(wv_w, wvb, (int)nw);
    cvt_f32_bf16<<<(int)((nw + 255) / 256), 256, 0, stream>>>(wo_w, wob, (int)nw);

    // 2) Q = x*Wq^T, K = x*Wk^T  (row-major [S,768], bf16)
    const int gemmBlocks = ((S_LEN / 64) * (D_MODEL / 64) + 7) / 8;   // 96
    gemm_bf16_nt<<<gemmBlocks, 256, 0, stream>>>(xb, wqb, Qb, wq_b,
                                                 S_LEN, D_MODEL, D_MODEL, 0);
    gemm_bf16_nt<<<gemmBlocks, 256, 0, stream>>>(xb, wkb, Kb, wk_b,
                                                 S_LEN, D_MODEL, D_MODEL, 0);
    // 3) V^T = Wv * x^T  ([768, S], bf16) -> same kernel, A/B swapped, bias by row
    gemm_bf16_nt<<<gemmBlocks, 256, 0, stream>>>(wvb, xb, Vtb, wv_b,
                                                 D_MODEL, S_LEN, D_MODEL, 2);

    // 4) flash attention: 12 heads x 16 q-blocks = 192 blocks of 8 waves
    attn_flash<<<N_HEADS * (S_LEN / 256), 256, 0, stream>>>(Qb, Kb, Vtb, ctxb);

    // 5) out = ctx * Wo^T + bo  (f32 output into d_out)
    gemm_bf16_nt<<<gemmBlocks, 256, 0, stream>>>(ctxb, wob, (float*)d_out, wo_b,
                                                 S_LEN, D_MODEL, D_MODEL, 1);
}
